// MultiHeadAttentionClass_51221779972556
// MI455X (gfx1250) — matrix operand order repaired and checked by reading, not compiled, not run on hardware
//
#include <hip/hip_runtime.h>
#include <hip/hip_bf16.h>

// Problem constants (from reference): B=512, T=128 (==BLOCK_SIZE), C=384, H=6, D=64
#define BB 512
#define TT 128
#define CC 384
#define HH 6
#define DD 64

#define XN   (BB * TT * CC)     // 25,165,824 elements
#define WN   (HH * CC * DD)     //    147,456 elements (per projection)
#define PN   (CC * CC)          //    147,456 elements
#define VLD  136                // padded LDS stride for V^T (avoids bank conflicts)

typedef _Float16 h16 __attribute__((ext_vector_type(16)));
typedef float    f8  __attribute__((ext_vector_type(8)));

// ---------------------------------------------------------------------------
// WMMA wrapper: D = A(16x32 f16) * B(32x16 f16) + C(16x16 f32)
// ---------------------------------------------------------------------------
static __device__ __forceinline__ f8 wmma16(h16 a, h16 b, f8 c) {
  return __builtin_amdgcn_wmma_f32_16x16x32_f16(
      /*neg_a=*/false, a, /*neg_b=*/false, b,
      /*c_mod=*/(short)0, c, /*reuse_a=*/false, /*reuse_b=*/false);
}

// ---------------------------------------------------------------------------
// Fragment loaders — wave32; A and B share a contiguous K order (not the ISA table order).
// A 16x32 f16: lane m=lane&15; lanes0-15 K={0..15}, lanes16-31 K={16..31} (contiguous)
// B 32x16 f16: lane n=lane&15; elem e = K row (kb+e), kb=0/16 by lane half
// C 16x16 f32: VGPR r: lanes0-15 = (row r, col lane), lanes16-31 = (row r+8)
// ---------------------------------------------------------------------------
static __device__ __forceinline__ h16 load_a_f16(const _Float16* __restrict__ src,
                                                 int ld, int lane) {
  const int m  = lane & 15;
  const int kb = (lane >> 4) << 4;               // 0 or 16
  const _Float16* p = src + (size_t)m * ld + kb;
  h16 a;
#pragma unroll
  for (int i = 0; i < 8; ++i) {
    a[i]     = p[i];                             // K = kb + i
    a[i + 8] = p[i + 8];                         // K = kb + 8 + i
  }
  return a;
}

// B fragment from a ROW-MAJOR-TRANSPOSED f16 source: logical B(kk,n) = src[n*ld + kk]
// -> per-lane contiguous 16-f16 (32B) read. Used for K^T, W^T, V^T, Wp^T.
static __device__ __forceinline__ h16 load_bT_f16(const _Float16* __restrict__ src,
                                                  int ld, int lane) {
  const int n  = lane & 15;
  const int kb = (lane >> 4) << 4;               // 0 or 16
  const _Float16* p = src + (size_t)n * ld + kb;
  h16 b;
#pragma unroll
  for (int e = 0; e < 16; ++e) b[e] = p[e];      // row kb+e, col n
  return b;
}

static __device__ __forceinline__ void store_c_f16(_Float16* __restrict__ dst,
                                                   int ld, const f8& c, int lane) {
  const int col = lane & 15;
  const int rb  = (lane >> 4) << 3;
#pragma unroll
  for (int r = 0; r < 8; ++r)
    dst[(size_t)(rb + r) * ld + col] = (_Float16)c[r];
}

// One 16x64 strip of x @ W : 12 K-chunks x 4 N-tiles of WMMA.
// WT is (D, C) so B fragments are contiguous; pointer comes straight from a
// kernel argument so these lower to global_load_b128 (LOADcnt-only waits).
static __device__ __forceinline__ void gemm_qkv(const _Float16* __restrict__ xb,
                                                const _Float16* __restrict__ WT,
                                                int m0, int lane, f8 (&c)[4]) {
  for (int kc = 0; kc < CC / 32; ++kc) {
    h16 a = load_a_f16(xb + (size_t)m0 * CC + kc * 32, CC, lane);
#pragma unroll
    for (int t = 0; t < 4; ++t) {
      h16 bf = load_bT_f16(WT + (size_t)(t * 16) * CC + kc * 32, CC, lane);
      c[t] = wmma16(a, bf, c[t]);
    }
  }
}

// ---------------------------------------------------------------------------
// Prep kernels (run once, graph-captured): f32 -> f16 convert (+ transpose).
// ---------------------------------------------------------------------------
__global__ __launch_bounds__(256)
void cvt_f32_f16(const float* __restrict__ in, _Float16* __restrict__ out, int n) {
  int i = blockIdx.x * blockDim.x + threadIdx.x;
  if (i < n) out[i] = (_Float16)in[i];
}

// (H, C, D) -> (H, D, C)
__global__ __launch_bounds__(256)
void cvtT_w(const float* __restrict__ in, _Float16* __restrict__ out) {
  int i = blockIdx.x * blockDim.x + threadIdx.x;
  if (i >= WN) return;
  int d = i % DD;
  int c = (i / DD) % CC;
  int h = i / (CC * DD);
  out[((size_t)h * DD + d) * CC + c] = (_Float16)in[i];
}

// (H*D, C) -> (C, H*D)
__global__ __launch_bounds__(256)
void cvtT_p(const float* __restrict__ in, _Float16* __restrict__ out) {
  int i = blockIdx.x * blockDim.x + threadIdx.x;
  if (i >= PN) return;
  int n = i % CC;
  int k = i / CC;
  out[(size_t)n * CC + k] = (_Float16)in[i];
}

// ---------------------------------------------------------------------------
// Kernel 1: per (b,h) fused QKV + causal softmax attention.
// 256 threads = 8 waves; wave w owns token rows [16w, 16w+16).
// x16: (B,T,C) f16.  wT: (H,D,C) f16.  att[b][t][h*64+d] f16 out.
// ---------------------------------------------------------------------------
__global__ __launch_bounds__(256)
void mha_attn_kernel(const _Float16* __restrict__ x16,
                     const _Float16* __restrict__ wqT,
                     const _Float16* __restrict__ wkT,
                     const _Float16* __restrict__ wvT,
                     _Float16* __restrict__ att) {
  __shared__ __align__(16) _Float16 lds[TT * DD * 2 + DD * VLD];  // ~50 KB
  _Float16* qs   = lds;                  // 128x64 row-major
  _Float16* ks   = lds + TT * DD;        // 128x64 row-major
  _Float16* vsT  = lds + 2 * TT * DD;    // 64x136 (V transposed, padded)
  _Float16* weis = lds;                  // 128x128, aliases qs+ks after they die

  const int b = blockIdx.x / HH;
  const int h = blockIdx.x % HH;
  const int lane = threadIdx.x & 31;
  const int m0 = (threadIdx.x >> 5) * 16;
  const int col0 = lane & 15;
  const int rb   = (lane >> 4) << 3;

  const _Float16* xb = x16 + (size_t)b * TT * CC;
  const size_t hoff = (size_t)h * DD * CC;

  // ---- Q = x @ Wq ----
  {
    f8 c[4] = {};
    gemm_qkv(xb, wqT + hoff, m0, lane, c);
#pragma unroll
    for (int t = 0; t < 4; ++t)
      store_c_f16(qs + (size_t)m0 * DD + t * 16, DD, c[t], lane);
  }
  // ---- K = x @ Wk ----
  {
    f8 c[4] = {};
    gemm_qkv(xb, wkT + hoff, m0, lane, c);
#pragma unroll
    for (int t = 0; t < 4; ++t)
      store_c_f16(ks + (size_t)m0 * DD + t * 16, DD, c[t], lane);
  }
  // ---- V = x @ Wv, stored transposed: vsT[d][s] (padded stride) ----
  {
    f8 c[4] = {};
    gemm_qkv(xb, wvT + hoff, m0, lane, c);
#pragma unroll
    for (int t = 0; t < 4; ++t)
#pragma unroll
      for (int r = 0; r < 8; ++r)
        vsT[(size_t)(t * 16 + col0) * VLD + m0 + rb + r] = (_Float16)c[t][r];
  }
  __syncthreads();

  // ---- scores = Q @ K^T (16x64 strip * 64x128) ----
  f8 s[8] = {};
  for (int kc = 0; kc < DD / 32; ++kc) {
    h16 a = load_a_f16(qs + (size_t)m0 * DD + kc * 32, DD, lane);
#pragma unroll
    for (int t = 0; t < 8; ++t) {
      h16 bf = load_bT_f16(ks + (size_t)(t * 16) * DD + kc * 32, DD, lane);
      s[t] = wmma16(a, bf, s[t]);
    }
  }

  // ---- scale + causal mask + softmax, fully in registers ----
  const float scale = 0.125f;   // D^-0.5
#pragma unroll
  for (int r = 0; r < 8; ++r) {
    const int row = m0 + rb + r;
    float mx = -INFINITY;
#pragma unroll
    for (int t = 0; t < 8; ++t) {
      const int col = t * 16 + col0;
      float v = s[t][r] * scale;
      v = (col <= row) ? v : -INFINITY;
      s[t][r] = v;
      mx = fmaxf(mx, v);
    }
#pragma unroll
    for (int off = 1; off < 16; off <<= 1)
      mx = fmaxf(mx, __shfl_xor(mx, off, 32));
    float sum = 0.0f;
#pragma unroll
    for (int t = 0; t < 8; ++t) {
      float p = __expf(s[t][r] - mx);
      s[t][r] = p;
      sum += p;
    }
#pragma unroll
    for (int off = 1; off < 16; off <<= 1)
      sum += __shfl_xor(sum, off, 32);
    const float inv = 1.0f / sum;
#pragma unroll
    for (int t = 0; t < 8; ++t) s[t][r] *= inv;
  }

  __syncthreads();   // everyone done reading qs/ks before weis overwrites them
#pragma unroll
  for (int t = 0; t < 8; ++t)
#pragma unroll
    for (int r = 0; r < 8; ++r)
      weis[(size_t)(m0 + rb + r) * TT + t * 16 + col0] = (_Float16)s[t][r];
  __syncthreads();

  // ---- O = wei @ V (16x128 strip * 128x64), V read transposed from LDS ----
  f8 o[4] = {};
  for (int kc = 0; kc < TT / 32; ++kc) {
    h16 a = load_a_f16(weis + (size_t)m0 * TT + kc * 32, TT, lane);
#pragma unroll
    for (int t = 0; t < 4; ++t) {
      h16 bf = load_bT_f16(vsT + (size_t)(t * 16) * VLD + kc * 32, VLD, lane);
      o[t] = wmma16(a, bf, o[t]);
    }
  }

  // att[b][t][h*64 + d], f16
#pragma unroll
  for (int t = 0; t < 4; ++t)
#pragma unroll
    for (int r = 0; r < 8; ++r)
      att[((size_t)b * TT + m0 + rb + r) * (HH * DD) + h * DD + t * 16 + col0] =
          (_Float16)o[t][r];
}

// ---------------------------------------------------------------------------
// Kernel 2: out = att (65536x384, f16) @ Wp + bp, fp32 out. WpT: (C, H*D) f16.
// grid (512, 6): 128-row x 64-col tile per workgroup; 8 waves of 16 rows.
// ---------------------------------------------------------------------------
__global__ __launch_bounds__(256)
void mha_proj_kernel(const _Float16* __restrict__ att,
                     const _Float16* __restrict__ wpT,
                     const float* __restrict__ bp,
                     float* __restrict__ out) {
  const int lane = threadIdx.x & 31;
  const int r0 = blockIdx.x * 128 + (threadIdx.x >> 5) * 16;
  const int n0 = blockIdx.y * 64;

  f8 c[4] = {};
  for (int kc = 0; kc < CC / 32; ++kc) {
    h16 a = load_a_f16(att + (size_t)r0 * CC + kc * 32, CC, lane);
#pragma unroll
    for (int t = 0; t < 4; ++t) {
      h16 bf = load_bT_f16(wpT + (size_t)(n0 + t * 16) * CC + kc * 32, CC, lane);
      c[t] = wmma16(a, bf, c[t]);
    }
  }

  const int col0 = lane & 15;
  const int rb   = (lane >> 4) << 3;
#pragma unroll
  for (int t = 0; t < 4; ++t) {
    const float bias = bp[n0 + t * 16 + col0];
#pragma unroll
    for (int r = 0; r < 8; ++r)
      out[((size_t)r0 + rb + r) * CC + n0 + t * 16 + col0] = c[t][r] + bias;
  }
}

// ---------------------------------------------------------------------------
extern "C" void kernel_launch(void* const* d_in, const int* in_sizes, int n_in,
                              void* d_out, int out_size, void* d_ws, size_t ws_size,
                              hipStream_t stream) {
  const float* x  = (const float*)d_in[0];   // (B, T, C)
  const float* Wq = (const float*)d_in[1];   // (H, C, D)
  const float* Wk = (const float*)d_in[2];   // (H, C, D)
  const float* Wv = (const float*)d_in[3];   // (H, C, D)
  const float* Wp = (const float*)d_in[4];   // (H*D, C)
  const float* bp = (const float*)d_in[5];   // (C,)
  float* out = (float*)d_out;                // (B, T, C) fp32

  // workspace layout (f16 elements)
  _Float16* att = (_Float16*)d_ws;           // XN
  _Float16* x16 = att + (size_t)XN;          // XN
  _Float16* wqT = x16 + (size_t)XN;          // WN  (H, D, C)
  _Float16* wkT = wqT + (size_t)WN;          // WN
  _Float16* wvT = wkT + (size_t)WN;          // WN
  _Float16* wpT = wvT + (size_t)WN;          // PN  (C, H*D)

  // one-time precision/layout prep
  cvt_f32_f16<<<dim3((XN + 255) / 256), dim3(256), 0, stream>>>(x, x16, XN);
  cvtT_w<<<dim3((WN + 255) / 256), dim3(256), 0, stream>>>(Wq, wqT);
  cvtT_w<<<dim3((WN + 255) / 256), dim3(256), 0, stream>>>(Wk, wkT);
  cvtT_w<<<dim3((WN + 255) / 256), dim3(256), 0, stream>>>(Wv, wvT);
  cvtT_p<<<dim3((PN + 255) / 256), dim3(256), 0, stream>>>(Wp, wpT);

  mha_attn_kernel<<<dim3(BB * HH), dim3(256), 0, stream>>>(x16, wqT, wkT, wvT, att);
  mha_proj_kernel<<<dim3(BB, CC / 64), dim3(256), 0, stream>>>(att, wpT, bp, out);
}
